// SelfAttention_21749714386967
// MI455X (gfx1250) — matrix operand order repaired and checked by reading, not compiled, not run on hardware
//
#include <hip/hip_runtime.h>
#include <cstdint>

// ---------------------------------------------------------------------------
// Self-attention for MI455X (gfx1250): bf16 WMMA everywhere, f32 accumulate.
//   x:[2,2048,1024] -> qkv GEMM -> flash-attention (TDM-staged K/V, online
//   softmax) -> proj.
// ---------------------------------------------------------------------------

typedef __attribute__((ext_vector_type(16))) __bf16 v16bf;
typedef __attribute__((ext_vector_type(8)))  __bf16 v8bf;
typedef __attribute__((ext_vector_type(8)))  float  v8f;
typedef __attribute__((ext_vector_type(4)))  unsigned int v4u;
typedef __attribute__((ext_vector_type(8)))  int v8i;
typedef __attribute__((ext_vector_type(4)))  int v4i;

#define DIM     1024
#define NHEADS  16
#define HDIM    64
#define SEQ     2048
#define BATCH   2
#define TOKENS  (BATCH * SEQ)           // 4096
// softmax scale folded with log2(e) so we can use exp2 (v_exp_f32)
#define SCL     (0.125f * 1.44269504088896340736f)

__device__ __forceinline__ unsigned short f2bfu(float f) {
  unsigned u = __builtin_bit_cast(unsigned, f);
  u += 0x7fffu + ((u >> 16) & 1u);          // round-to-nearest-even
  return (unsigned short)(u >> 16);
}
__device__ __forceinline__ __bf16 f2bf(float f) {
  unsigned short s = f2bfu(f);
  return __builtin_bit_cast(__bf16, s);
}
// cross-lane move via LDS-permute hardware (no LDS memory touched); wave32
__device__ __forceinline__ float permf(float v, int srcLane) {
  int i = __builtin_bit_cast(int, v);
  i = __builtin_amdgcn_ds_bpermute(srcLane << 2, i);
  return __builtin_bit_cast(float, i);
}
__device__ __forceinline__ v8f wmma_bf16(v16bf a, v16bf b, v8f c) {
  // (neg_a, A, neg_b, B, c_mod, C, reuse_a, reuse_b)
  return __builtin_amdgcn_wmma_f32_16x16x32_bf16(false, a, false, b, (short)0, c,
                                                 false, false);
}
// A-fragment (16x32) in this file's contiguous k order (B loads match): lanes 0-15 hold K 0-15, lanes 16-31
// hold K 16-31 of row M=lane&15.  Two 16B loads per lane.
__device__ __forceinline__ v16bf load_a(const unsigned short* row, int elemBase,
                                        int lh) {
  v8bf lo = *(const v8bf*)(row + elemBase + lh * 16);
  v8bf hi = *(const v8bf*)(row + elemBase + lh * 16 + 8);
  return __builtin_shufflevector(lo, hi, 0, 1, 2, 3, 4, 5, 6, 7, 8, 9, 10, 11,
                                 12, 13, 14, 15);
}

// ---- Tensor Data Mover: async 2-D bf16 tile load Global->LDS --------------
// Descriptor layout per CDNA5 ISA 08_async_tensor sec.8 (D# groups 0..3).
// dims/strides in data_size units (2-byte elements); groups 2/3 zero (2-D).
// This toolchain's builtin is the 6-arg form:
//   (v4u g0, v8i g1, v4i g2, v4i g3, v8i, i32 cpol)
__device__ __forceinline__ void tdm_load_2d(unsigned lds_off,
                                            unsigned long long gaddr,
                                            unsigned tensor_d0,
                                            unsigned tensor_d1,
                                            unsigned tile_d0, unsigned tile_d1,
                                            unsigned long long stride0) {
  v4u g0;
  g0[0] = 1u;                                    // count=1, user descriptor
  g0[1] = lds_off;                               // lds_addr (bytes)
  g0[2] = (unsigned)(gaddr & 0xffffffffu);       // global_addr[31:0]
  g0[3] = (unsigned)((gaddr >> 32) & 0x01ffffffu) | (2u << 30);  // type=2
  v8i g1;
  g1[0] = (int)(1u << 16);                       // data_size=1 (2 bytes)
  g1[1] = (int)((tensor_d0 & 0xffffu) << 16);    // tensor_dim0[15:0] @ 63:48
  g1[2] = (int)(((tensor_d0 >> 16) & 0xffffu) |
                ((tensor_d1 & 0xffffu) << 16));  // dim0[31:16] | dim1[15:0]
  g1[3] = (int)(((tensor_d1 >> 16) & 0xffffu) |
                ((tile_d0 & 0xffffu) << 16));    // dim1[31:16] | tile_dim0
  g1[4] = (int)(tile_d1 & 0xffffu);              // tile_dim1 | tile_dim2=0
  g1[5] = (int)(unsigned)(stride0 & 0xffffffffu);        // dim0_stride[31:0]
  g1[6] = (int)(unsigned)((stride0 >> 32) & 0xffffu);    // [47:32] | dim1_str
  g1[7] = 0;
  v4i z4 = {0, 0, 0, 0};
  v8i z8 = {0, 0, 0, 0, 0, 0, 0, 0};
  __builtin_amdgcn_tensor_load_to_lds(g0, g1, z4, z4, z8, 0);
}

// --------------------------- fp32 -> bf16 ----------------------------------
__global__ void cvt_bf16_kernel(const float* __restrict__ in,
                                unsigned short* __restrict__ out, int n) {
  int i = blockIdx.x * blockDim.x + threadIdx.x;
  int stride = gridDim.x * blockDim.x;
  for (; i < n; i += stride) out[i] = f2bfu(in[i]);
}

// ----------------------- QKV projection GEMM -------------------------------
// [4096,1024]bf16 @ [3072,1024]^T bf16 -> scatter into per-head Q,K,V(T).
// One wave computes a 16x64 output tile: 32 K-chunks x 4 WMMAs.
__global__ __launch_bounds__(256) void qkv_gemm_kernel(
    const unsigned short* __restrict__ A, const unsigned short* __restrict__ W,
    unsigned short* __restrict__ Qb, unsigned short* __restrict__ Kb,
    unsigned short* __restrict__ Vt) {
  const int lane = threadIdx.x & 31;
  const int col = lane & 15, lh = lane >> 4;
  const int w = blockIdx.x * 8 + (threadIdx.x >> 5);  // 12288 waves = 256*48
  const int mt = w / 48, dt = w % 48;

  const unsigned short* arow = A + (size_t)(mt * 16 + col) * DIM;
  const unsigned short* wbase = W + (size_t)dt * 64 * DIM;
  v8f acc[4] = {};
  for (int kc = 0; kc < 32; ++kc) {
    v16bf a = load_a(arow, kc * 32, lh);
#pragma unroll
    for (int g = 0; g < 4; ++g) {
      // B col n = W row (output channel); lanes 0-15 K=0-15, 16-31 K=16-31
      const unsigned short* wp =
          wbase + (size_t)(g * 16 + col) * DIM + kc * 32 + lh * 16;
      v16bf b = *(const v16bf*)wp;
      acc[g] = wmma_bf16(a, b, acc[g]);
    }
  }
  // scatter: d<1024 -> Q[bh][n][hd]; <2048 -> K; else V transposed [bh][hd][n]
#pragma unroll
  for (int g = 0; g < 4; ++g) {
    int d = dt * 64 + g * 16 + col;
    int sec = d >> 10, dd = d & 1023, h = dd >> 6, hd = dd & 63;
#pragma unroll
    for (int r = 0; r < 8; ++r) {
      int token = mt * 16 + 8 * lh + r;      // C row = r + 8*(lane>>4)
      int b_ = token >> 11, ns = token & 2047;
      unsigned short bv = f2bfu(acc[g][r]);
      size_t bh = (size_t)(b_ * NHEADS + h);
      if (sec == 0)      Qb[(bh * SEQ + ns) * HDIM + hd] = bv;
      else if (sec == 1) Kb[(bh * SEQ + ns) * HDIM + hd] = bv;
      else               Vt[(bh * HDIM + hd) * SEQ + ns] = bv;
    }
  }
}

// ------------------------- flash attention core ----------------------------
// One wave owns a 16-query tile; workgroup (4 waves, same batch*head) streams
// 2048 keys in 32-key blocks.  K (32x64) and V^T (64x32) blocks are staged
// into LDS by the Tensor Data Mover, double-buffered: wave 0 issues the DMA
// for block i+1 while all waves compute block i from LDS; s_wait_tensorcnt +
// barrier publishes the next buffer.  Computes S^T = K.Q^T so the softmaxed
// C-fragment maps directly onto the A-fragment for P.V (interleaved-by-8
// CDNA5 A layout) -- no cross-lane shuffle for the P operand.
__global__ __launch_bounds__(128) void attn_kernel(
    const unsigned short* __restrict__ Q, const unsigned short* __restrict__ K,
    const unsigned short* __restrict__ VT, unsigned short* __restrict__ AO) {
  __shared__ unsigned short kS[2][32 * HDIM];   // 2 x 4KB
  __shared__ unsigned short vS[2][HDIM * 32];   // 2 x 4KB

  const int lane = threadIdx.x & 31;
  const int col = lane & 15, lh = lane >> 4;
  const int wv = threadIdx.x >> 5;
  const int bh = blockIdx.x >> 5;                    // 0..31 (batch*head)
  const int qt = (blockIdx.x & 31) * 64 + wv * 16;   // query tile start
  const int b_ = bh >> 4, h = bh & 15;

  const unsigned short* Qh = Q + (size_t)bh * SEQ * HDIM;
  const unsigned short* Kh = K + (size_t)bh * SEQ * HDIM;
  const unsigned short* Vh = VT + (size_t)bh * HDIM * SEQ;

  // Q^T as B-fragments (column n = query, rows = head-dim), 2 chunks of 32
  const unsigned short* qrow = Qh + (size_t)(qt + col) * HDIM;
  v16bf qb0 = *(const v16bf*)(qrow + lh * 16);        // hd 0..31
  v16bf qb1 = *(const v16bf*)(qrow + 32 + lh * 16);   // hd 32..63

  const unsigned kOff0 = (unsigned)(size_t)(void*)&kS[0][0];
  const unsigned kOff1 = (unsigned)(size_t)(void*)&kS[1][0];
  const unsigned vOff0 = (unsigned)(size_t)(void*)&vS[0][0];
  const unsigned vOff1 = (unsigned)(size_t)(void*)&vS[1][0];

  // prologue: DMA block 0 into buffer 0
  if (wv == 0) {
    tdm_load_2d(kOff0, (unsigned long long)(size_t)Kh,
                HDIM, SEQ, HDIM, 32, HDIM);
    tdm_load_2d(vOff0, (unsigned long long)(size_t)Vh,
                SEQ, HDIM, 32, HDIM, SEQ);
  }
  __builtin_amdgcn_s_wait_tensorcnt(0);
  __syncthreads();

  v8f o[4] = {};                 // 16x64 output accumulator (4 C-frags)
  float m2 = -1.0e30f;           // running row max (exp2 domain)
  float l = 0.0f;                // running row sum

  for (int kb = 0; kb < SEQ; kb += 32) {
    const int p = (kb >> 5) & 1;
    // issue DMA for the NEXT 32-key block into the other buffer (that buffer
    // was consumed in the previous iteration; the trailing barrier of that
    // iteration guarantees all waves are done reading it)
    if (wv == 0 && kb + 32 < SEQ) {
      tdm_load_2d(p ? kOff0 : kOff1,
                  (unsigned long long)(size_t)(Kh + (size_t)(kb + 32) * HDIM),
                  HDIM, SEQ, HDIM, 32, HDIM);
      tdm_load_2d(p ? vOff0 : vOff1,
                  (unsigned long long)(size_t)(Vh + (kb + 32)),
                  SEQ, HDIM, 32, HDIM, SEQ);
    }

    // ---- S^T for 32 keys from LDS: two 16x16 tiles, K-dim=64 in two WMMAs
    const unsigned short* kt = &kS[p][0];        // [32 keys][64 hd]
    v8f c0 = {}, c1 = {};
    c0 = wmma_bf16(load_a(kt + col * HDIM, 0, lh), qb0, c0);
    c0 = wmma_bf16(load_a(kt + col * HDIM, 32, lh), qb1, c0);
    c1 = wmma_bf16(load_a(kt + (col + 16) * HDIM, 0, lh), qb0, c1);
    c1 = wmma_bf16(load_a(kt + (col + 16) * HDIM, 32, lh), qb1, c1);

    // ---- online softmax (per lane: 16 of 32 scores of query `col`;
    //      the other 16 live in lane^16)
    float bm = -3.0e38f;
#pragma unroll
    for (int i = 0; i < 8; ++i) {
      c0[i] *= SCL; c1[i] *= SCL;
      bm = fmaxf(bm, fmaxf(c0[i], c1[i]));
    }
    bm = fmaxf(bm, permf(bm, lane ^ 16));
    float mnew = fmaxf(m2, bm);
    float corr = exp2f(m2 - mnew);
    m2 = mnew;

    float rs = 0.0f;
    v16bf pA;                       // P as A-fragment: direct C->A mapping
#pragma unroll
    for (int i = 0; i < 8; ++i) {
      float p0 = exp2f(c0[i] - mnew); rs += p0; pA[i] = f2bf(p0);
    }
#pragma unroll
    for (int i = 0; i < 8; ++i) {
      float p1 = exp2f(c1[i] - mnew); rs += p1; pA[8 + i] = f2bf(p1);
    }
    rs += permf(rs, lane ^ 16);
    l = l * corr + rs;

    // rescale accumulator rows (row q = 8*lh + r; its corr lives in lane q)
#pragma unroll
    for (int r = 0; r < 8; ++r) {
      float s = permf(corr, (lh << 3) + r);
      o[0][r] *= s; o[1][r] *= s; o[2][r] *= s; o[3][r] *= s;
    }

    // ---- P.V : B-fragment = V^T tile rows in LDS (contiguous keys)
    const unsigned short* vt = &vS[p][0];        // [64 hd][32 keys]
#pragma unroll
    for (int g = 0; g < 4; ++g) {
      const unsigned short* vr = vt + (g * 16 + col) * 32;
      v16bf vb = __builtin_shufflevector(*(const v8bf*)(vr + lh * 8),
                                         *(const v8bf*)(vr + 16 + lh * 8), 0, 1,
                                         2, 3, 4, 5, 6, 7, 8, 9, 10, 11, 12, 13,
                                         14, 15);
      o[g] = wmma_bf16(pA, vb, o[g]);
    }

    // publish next buffer / retire this one
    __builtin_amdgcn_s_wait_tensorcnt(0);
    __syncthreads();
  }

  // ---- normalize rows and store bf16 into [token][DIM] for proj GEMM
#pragma unroll
  for (int r = 0; r < 8; ++r) {
    float li = permf(l, (lh << 3) + r);
    float inv = 1.0f / li;
    size_t base =
        (size_t)(b_ * SEQ + qt + 8 * lh + r) * DIM + h * HDIM + col;
#pragma unroll
    for (int g = 0; g < 4; ++g) AO[base + g * 16] = f2bfu(o[g][r] * inv);
  }
}

// --------------------------- output projection -----------------------------
__global__ __launch_bounds__(256) void proj_gemm_kernel(
    const unsigned short* __restrict__ A, const unsigned short* __restrict__ W,
    const float* __restrict__ bias, float* __restrict__ out) {
  const int lane = threadIdx.x & 31;
  const int col = lane & 15, lh = lane >> 4;
  const int w = blockIdx.x * 8 + (threadIdx.x >> 5);  // 4096 waves = 256*16
  const int mt = w >> 4, nt = w & 15;

  const unsigned short* arow = A + (size_t)(mt * 16 + col) * DIM;
  const unsigned short* wbase = W + (size_t)nt * 64 * DIM;
  v8f acc[4] = {};
  for (int kc = 0; kc < 32; ++kc) {
    v16bf a = load_a(arow, kc * 32, lh);
#pragma unroll
    for (int g = 0; g < 4; ++g) {
      const unsigned short* wp =
          wbase + (size_t)(g * 16 + col) * DIM + kc * 32 + lh * 16;
      v16bf b = *(const v16bf*)wp;
      acc[g] = wmma_bf16(a, b, acc[g]);
    }
  }
#pragma unroll
  for (int g = 0; g < 4; ++g) {
    int c = nt * 64 + g * 16 + col;
    float bv = bias[c];
#pragma unroll
    for (int r = 0; r < 8; ++r) {
      out[(size_t)(mt * 16 + 8 * lh + r) * DIM + c] = acc[g][r] + bv;
    }
  }
}

// ---------------------------------------------------------------------------
extern "C" void kernel_launch(void* const* d_in, const int* in_sizes, int n_in,
                              void* d_out, int out_size, void* d_ws,
                              size_t ws_size, hipStream_t stream) {
  const float* x    = (const float*)d_in[0];  // [2,2048,1024]
  const float* qkvw = (const float*)d_in[1];  // [3072,1024]
  const float* pw   = (const float*)d_in[2];  // [1024,1024]
  const float* pb   = (const float*)d_in[3];  // [1024]

  // workspace carve (40 MB): xbf is dead after qkv GEMM -> reused for AO
  char* p = (char*)d_ws;
  unsigned short* xbf = (unsigned short*)p; p += (size_t)TOKENS * DIM * 2;
  unsigned short* wq  = (unsigned short*)p; p += (size_t)3 * DIM * DIM * 2;
  unsigned short* wp  = (unsigned short*)p; p += (size_t)DIM * DIM * 2;
  unsigned short* Qb  = (unsigned short*)p; p += (size_t)TOKENS * DIM * 2;
  unsigned short* Kb  = (unsigned short*)p; p += (size_t)TOKENS * DIM * 2;
  unsigned short* Vt  = (unsigned short*)p; p += (size_t)TOKENS * DIM * 2;
  unsigned short* AO  = xbf;

  cvt_bf16_kernel<<<2048, 256, 0, stream>>>(x, xbf, TOKENS * DIM);
  cvt_bf16_kernel<<<2048, 256, 0, stream>>>(qkvw, wq, 3 * DIM * DIM);
  cvt_bf16_kernel<<<1024, 256, 0, stream>>>(pw, wp, DIM * DIM);

  qkv_gemm_kernel<<<1536, 256, 0, stream>>>(xbf, wq, Qb, Kb, Vt);   // 12288 waves
  attn_kernel<<<1024, 128, 0, stream>>>(Qb, Kb, Vt, AO);            // 4096 waves
  proj_gemm_kernel<<<512, 256, 0, stream>>>(AO, wp, pb, (float*)d_out);
}
